// LearnedSimulator_154618822750
// MI455X (gfx1250) — matrix operand order repaired and checked by reading, not compiled, not run on hardware
//
#include <hip/hip_runtime.h>
#include <hip/hip_bf16.h>

// ---------------------------------------------------------------------------
// GNS learned-simulator forward for MI455X (gfx1250, wave32, WMMA).
// All GEMMs: bf16 inputs, fp32 accumulate via v_wmma_f32_16x16x32_bf16.
// One wave (32 lanes) owns one 16-row tile (16 edges or 16 nodes) end-to-end.
// ---------------------------------------------------------------------------

#define HID  128
#define LDSW 132   // 16-row fp32 tile stride (pad to dodge LDS bank conflicts)

typedef __attribute__((ext_vector_type(16))) __bf16 v16bf;
typedef __attribute__((ext_vector_type(8)))  __bf16 v8bf;
typedef __attribute__((ext_vector_type(4)))  __bf16 v4bf;
typedef __attribute__((ext_vector_type(8)))  float  v8f;
typedef __attribute__((ext_vector_type(4)))  float  v4f;

// Same-wave LDS producer->consumer fence (single-wave workgroups, no barrier).
#define LDS_SYNC() asm volatile("s_wait_dscnt 0" ::: "memory")

__device__ __forceinline__ v8f wmma_bf16(v16bf a, v16bf b, v8f c) {
  return __builtin_amdgcn_wmma_f32_16x16x32_bf16(false, a, false, b,
                                                 (short)0, c, false, false);
}

// A-operand (16x32 bf16) for one K-tile from a bf16 row in memory.
// ISA layout: lanes 0-15 hold K {kt*32+0..7, +16..23}; lanes 16-31 {+8..15, +24..31}.
__device__ __forceinline__ v16bf a_from_bf16row(const __bf16* row, int kt, int lane) {
  int klo = kt * 32 + ((lane & 16) ? 8 : 0);
  v8bf lo = *(const v8bf*)(row + klo);
  v8bf hi = *(const v8bf*)(row + klo + 16);
  return __builtin_shufflevector(lo, hi, 0,1,2,3,4,5,6,7,8,9,10,11,12,13,14,15);
}

// A-operand from an fp32 row (global or LDS), converting to bf16 in-register.
__device__ __forceinline__ v16bf a_from_f32row(const float* row, int kt, int lane) {
  int klo = kt * 32 + ((lane & 16) ? 8 : 0);
  v4f x0 = *(const v4f*)(row + klo);
  v4f x1 = *(const v4f*)(row + klo + 4);
  v4f x2 = *(const v4f*)(row + klo + 16);
  v4f x3 = *(const v4f*)(row + klo + 20);
  v16bf a;
#pragma unroll
  for (int j = 0; j < 4; ++j) {
    a[j]      = (__bf16)x0[j];
    a[4 + j]  = (__bf16)x1[j];
    a[8 + j]  = (__bf16)x2[j];
    a[12 + j] = (__bf16)x3[j];
  }
  return a;
}

// B-operand from pre-packed weights: tile (kt,nt) stores 32 lanes x 16 bf16.
__device__ __forceinline__ v16bf load_b(const __bf16* Wp, int kt, int nt, int Nt, int lane) {
  return *(const v16bf*)(Wp + ((size_t)(kt * Nt + nt) * 32 + lane) * 16);
}

// D(16x128) + bias (+ReLU) -> fp32 LDS tile (row-major, stride LDSW).
__device__ __forceinline__ void store_tile(float* sOut, const v8f* acc,
                                           const float* bias, int lane, bool relu) {
  int col = lane & 15;
  int rb  = (lane & 16) ? 8 : 0;
#pragma unroll
  for (int nt = 0; nt < 8; ++nt) {
    int   n  = nt * 16 + col;
    float bv = bias[n];
#pragma unroll
    for (int r = 0; r < 8; ++r) {
      float v = acc[nt][r] + bv;
      if (relu) v = fmaxf(v, 0.0f);
      sOut[(size_t)(rb + r) * LDSW + n] = v;
    }
  }
}

// One dense layer: LDS fp32 (16 x 32*Kt) x packed W (32*Kt x 128) -> LDS fp32.
__device__ __forceinline__ void dense_from_lds(const float* sIn, float* sOut,
                                               const __bf16* Wp, const float* bias,
                                               int Kt, int lane, bool relu) {
  v8f zero = {};
  v8f acc[8];
#pragma unroll
  for (int nt = 0; nt < 8; ++nt) acc[nt] = zero;
  const float* row = sIn + (size_t)(lane & 15) * LDSW;
  for (int kt = 0; kt < Kt; ++kt) {
    v16bf a = a_from_f32row(row, kt, lane);
#pragma unroll
    for (int nt = 0; nt < 8; ++nt)
      acc[nt] = wmma_bf16(a, load_b(Wp, kt, nt, 8, lane), acc[nt]);
  }
  store_tile(sOut, acc, bias, lane, relu);
}

// LayerNorm over each 128-wide row of the LDS tile (eps = 1e-5).
__device__ __forceinline__ void layernorm_tile(float* s, float* stats,
                                               const float* g, const float* b, int lane) {
  if (lane < 16) {
    float* row = s + (size_t)lane * LDSW;
    float mu = 0.f;
    for (int i = 0; i < HID; ++i) mu += row[i];
    mu *= (1.f / HID);
    float var = 0.f;
    for (int i = 0; i < HID; ++i) { float d = row[i] - mu; var += d * d; }
    var *= (1.f / HID);
    stats[lane * 2]     = mu;
    stats[lane * 2 + 1] = rsqrtf(var + 1e-5f);
  }
  LDS_SYNC();
  int m = lane & 15, c0 = (lane & 16) ? 64 : 0;
  float mu = stats[m * 2], rs = stats[m * 2 + 1];
  float* row = s + (size_t)m * LDSW;
  for (int i = 0; i < 64; ++i) {
    int n = c0 + i;
    row[n] = (row[n] - mu) * rs * g[n] + b[n];
  }
}

// ---------------------------------------------------------------------------
// Weight packing: row-major fp32 W[K][N] -> bf16 WMMA-B tiles.
// Tile (kt,nt): lane l holds K {kbase+0..7, kbase+16..23} (kbase = kt*32 + (l<16?0:8))
// of column n = nt*16 + (l&15). Zero-padded beyond K,N.
// ---------------------------------------------------------------------------
__global__ void pack_w_kernel(const float* __restrict__ W, __bf16* __restrict__ dst,
                              int K, int N, int tiles, int Nt) {
  int t    = blockIdx.x * blockDim.x + threadIdx.x;
  int tile = t >> 5, lane = t & 31;
  if (tile >= tiles) return;
  int kt = tile / Nt, nt = tile - kt * Nt;
  int n  = nt * 16 + (lane & 15);
  int kb = kt * 32 + ((lane & 16) ? 8 : 0);
  __bf16* o = dst + ((size_t)tile * 32 + lane) * 16;
#pragma unroll
  for (int i = 0; i < 16; ++i) {
    int k = kb + (i < 8 ? i : i + 8);
    float v = (k < K && n < N) ? W[(size_t)k * N + n] : 0.f;
    o[i] = (__bf16)v;
  }
}

__global__ void zero_f32_kernel(float* __restrict__ p, long n) {
  long i = (long)blockIdx.x * blockDim.x + threadIdx.x;
  if (i < n) p[i] = 0.f;
}

// ---------------------------------------------------------------------------
// Encoder: node_in MLP over [embed[ptype](16) | node_attr(14)] (pad K->32)
// ---------------------------------------------------------------------------
__global__ void encode_nodes_kernel(const int* __restrict__ ptype,
                                    const float* __restrict__ node_attr,
                                    const float* __restrict__ embed,
                                    const __bf16* Wp0, const __bf16* Wp1, const __bf16* Wp2,
                                    const float* b0, const float* b1, const float* b2,
                                    const float* lng, const float* lnb,
                                    float* __restrict__ h_f32, __bf16* __restrict__ h_bf,
                                    int nNodes) {
  __shared__ float sA[16 * LDSW];
  __shared__ float sB[16 * LDSW];
  __shared__ float stats[32];
  int lane = threadIdx.x, tile = blockIdx.x;
  if (lane < 16) {
    int node = tile * 16 + lane;
    if (node >= nNodes) node = nNodes - 1;
    int pt = ptype[node];
    float* row = sA + (size_t)lane * LDSW;
#pragma unroll
    for (int i = 0; i < 16; ++i) row[i] = embed[pt * 16 + i];
#pragma unroll
    for (int i = 0; i < 14; ++i) row[16 + i] = node_attr[(size_t)node * 14 + i];
    row[30] = 0.f; row[31] = 0.f;
  }
  LDS_SYNC();
  dense_from_lds(sA, sB, Wp0, b0, 1, lane, true);
  LDS_SYNC();
  dense_from_lds(sB, sA, Wp1, b1, 4, lane, true);
  LDS_SYNC();
  dense_from_lds(sA, sB, Wp2, b2, 4, lane, false);
  LDS_SYNC();
  layernorm_tile(sB, stats, lng, lnb, lane);
  LDS_SYNC();
  int m = lane & 15, c0 = (lane & 16) ? 64 : 0;
  int node = tile * 16 + m;
  if (node >= nNodes) return;
  const float* orow = sB + (size_t)m * LDSW;
  float*  hrow = h_f32 + (size_t)node * HID;
  __bf16* brow = h_bf  + (size_t)node * HID;
#pragma unroll
  for (int cb = 0; cb < 16; ++cb) {
    int n0 = c0 + cb * 4;
    v4f  hv; v4bf hb;
#pragma unroll
    for (int j = 0; j < 4; ++j) { float v = orow[n0 + j]; hv[j] = v; hb[j] = (__bf16)v; }
    *(v4f*) (hrow + n0) = hv;
    *(v4bf*)(brow + n0) = hb;
  }
}

// ---------------------------------------------------------------------------
// Encoder: edge_in MLP over edge_attr(3) (pad K->32) -> e_bf16
// ---------------------------------------------------------------------------
__global__ void encode_edges_kernel(const float* __restrict__ edge_attr,
                                    const __bf16* Wp0, const __bf16* Wp1, const __bf16* Wp2,
                                    const float* b0, const float* b1, const float* b2,
                                    const float* lng, const float* lnb,
                                    __bf16* __restrict__ e_bf, int nEdges) {
  __shared__ float sA[16 * LDSW];
  __shared__ float sB[16 * LDSW];
  __shared__ float stats[32];
  int lane = threadIdx.x, tile = blockIdx.x;
  if (lane < 16) {
    int edge = tile * 16 + lane;
    if (edge >= nEdges) edge = nEdges - 1;
    float* row = sA + (size_t)lane * LDSW;
#pragma unroll
    for (int i = 0; i < 32; ++i) row[i] = 0.f;
    row[0] = edge_attr[(size_t)edge * 3 + 0];
    row[1] = edge_attr[(size_t)edge * 3 + 1];
    row[2] = edge_attr[(size_t)edge * 3 + 2];
  }
  LDS_SYNC();
  dense_from_lds(sA, sB, Wp0, b0, 1, lane, true);
  LDS_SYNC();
  dense_from_lds(sB, sA, Wp1, b1, 4, lane, true);
  LDS_SYNC();
  dense_from_lds(sA, sB, Wp2, b2, 4, lane, false);
  LDS_SYNC();
  layernorm_tile(sB, stats, lng, lnb, lane);
  LDS_SYNC();
  int m = lane & 15, c0 = (lane & 16) ? 64 : 0;
  int edge = tile * 16 + m;
  if (edge >= nEdges) return;
  const float* orow = sB + (size_t)m * LDSW;
  __bf16* erow = e_bf + (size_t)edge * HID;
#pragma unroll
  for (int cb = 0; cb < 8; ++cb) {
    int n0 = c0 + cb * 8;
    v8bf ev;
#pragma unroll
    for (int j = 0; j < 8; ++j) ev[j] = (__bf16)orow[n0 + j];
    *(v8bf*)(erow + n0) = ev;
  }
}

// ---------------------------------------------------------------------------
// MP layer, edge half: msg = MLP([h[dst] | h[src] | e]); e += msg;
//                      aggr[dst] += msg * node_dist   (f32 atomics, L2-resident)
// ---------------------------------------------------------------------------
__global__ void edge_msg_kernel(const __bf16* __restrict__ h_bf,
                                __bf16* __restrict__ e_bf,
                                float* __restrict__ aggr,
                                const int* __restrict__ edge_index,
                                const float* __restrict__ node_dist,
                                const __bf16* Wp0, const __bf16* Wp1, const __bf16* Wp2,
                                const float* b0, const float* b1, const float* b2,
                                const float* lng, const float* lnb, int nEdges) {
  __shared__ float sA[16 * LDSW];
  __shared__ float sB[16 * LDSW];
  __shared__ float stats[32];
  int lane = threadIdx.x, tile = blockIdx.x;
  int m    = lane & 15;
  int edge = tile * 16 + m;
  bool valid = (edge < nEdges);
  if (!valid) edge = nEdges - 1;
  int src = edge_index[edge];
  int dst = edge_index[nEdges + edge];
  const __bf16* a0 = h_bf + (size_t)dst * HID;   // x_i = h[dst]
  const __bf16* a1 = h_bf + (size_t)src * HID;   // x_j = h[src]
  const __bf16* a2 = e_bf + (size_t)edge * HID;

  v8f zero = {};
  v8f acc[8];
#pragma unroll
  for (int nt = 0; nt < 8; ++nt) acc[nt] = zero;
  for (int kt = 0; kt < 12; ++kt) {               // K = 384
    const __bf16* base = (kt < 4) ? a0 : (kt < 8 ? a1 : a2);
    v16bf a = a_from_bf16row(base, kt & 3, lane);
#pragma unroll
    for (int nt = 0; nt < 8; ++nt)
      acc[nt] = wmma_bf16(a, load_b(Wp0, kt, nt, 8, lane), acc[nt]);
  }
  store_tile(sA, acc, b0, lane, true);
  LDS_SYNC();
  dense_from_lds(sA, sB, Wp1, b1, 4, lane, true);
  LDS_SYNC();
  dense_from_lds(sB, sA, Wp2, b2, 4, lane, false);
  LDS_SYNC();
  layernorm_tile(sA, stats, lng, lnb, lane);
  LDS_SYNC();

  if (!valid) return;
  float w = node_dist[edge];
  __bf16* erow = e_bf + (size_t)edge * HID;
  float*  arow = aggr + (size_t)dst  * HID;
  const float* mrow = sA + (size_t)m * LDSW;
  int c0 = (lane & 16) ? 64 : 0;
#pragma unroll
  for (int cb = 0; cb < 8; ++cb) {
    int n0 = c0 + cb * 8;
    v8bf ev = *(const v8bf*)(erow + n0);
    v8bf out;
#pragma unroll
    for (int j = 0; j < 8; ++j) {
      float msg = mrow[n0 + j];
      out[j] = (__bf16)((float)ev[j] + msg);
      unsafeAtomicAdd(arow + n0 + j, msg * w);   // global_atomic_add_f32
    }
    *(v8bf*)(erow + n0) = out;
  }
}

// ---------------------------------------------------------------------------
// MP layer, node half: h += MLP([h | aggr]); then zero aggr for next layer.
// ---------------------------------------------------------------------------
__global__ void node_update_kernel(float* __restrict__ h_f32,
                                   __bf16* __restrict__ h_bf,
                                   float* __restrict__ aggr,
                                   const __bf16* Wp0, const __bf16* Wp1, const __bf16* Wp2,
                                   const float* b0, const float* b1, const float* b2,
                                   const float* lng, const float* lnb, int nNodes) {
  __shared__ float sA[16 * LDSW];
  __shared__ float sB[16 * LDSW];
  __shared__ float stats[32];
  int lane = threadIdx.x, tile = blockIdx.x;
  int m    = lane & 15;
  int node = tile * 16 + m;
  bool valid = (node < nNodes);
  if (!valid) node = nNodes - 1;
  const __bf16* hrowb = h_bf + (size_t)node * HID;
  const float*  arowf = aggr + (size_t)node * HID;

  v8f zero = {};
  v8f acc[8];
#pragma unroll
  for (int nt = 0; nt < 8; ++nt) acc[nt] = zero;
  for (int kt = 0; kt < 8; ++kt) {                // K = 256 : [h | aggr]
    v16bf a = (kt < 4) ? a_from_bf16row(hrowb, kt, lane)
                       : a_from_f32row(arowf, kt - 4, lane);
#pragma unroll
    for (int nt = 0; nt < 8; ++nt)
      acc[nt] = wmma_bf16(a, load_b(Wp0, kt, nt, 8, lane), acc[nt]);
  }
  store_tile(sA, acc, b0, lane, true);
  LDS_SYNC();
  dense_from_lds(sA, sB, Wp1, b1, 4, lane, true);
  LDS_SYNC();
  dense_from_lds(sB, sA, Wp2, b2, 4, lane, false);
  LDS_SYNC();
  layernorm_tile(sA, stats, lng, lnb, lane);
  LDS_SYNC();

  if (!valid) return;
  const float* orow = sA + (size_t)m * LDSW;
  float*  hrow  = h_f32 + (size_t)node * HID;
  __bf16* hbrow = h_bf  + (size_t)node * HID;
  float*  arow  = aggr  + (size_t)node * HID;
  int c0 = (lane & 16) ? 64 : 0;
  v4f z = {};
#pragma unroll
  for (int cb = 0; cb < 16; ++cb) {
    int n0 = c0 + cb * 4;
    v4f hv = *(const v4f*)(hrow + n0);
    v4bf hb;
#pragma unroll
    for (int j = 0; j < 4; ++j) {
      float v = hv[j] + orow[n0 + j];
      hv[j] = v;
      hb[j] = (__bf16)v;
    }
    *(v4f*) (hrow + n0)  = hv;
    *(v4bf*)(hbrow + n0) = hb;
    *(v4f*) (arow + n0)  = z;      // pre-zero aggr for next layer
  }
}

// ---------------------------------------------------------------------------
// Decoder: node_out MLP 128 ->128(ReLU) ->128(ReLU) -> 2  (no LN)
// ---------------------------------------------------------------------------
__global__ void decode_kernel(const __bf16* __restrict__ h_bf,
                              const __bf16* Wp0, const __bf16* Wp1, const __bf16* Wp2,
                              const float* b0, const float* b1, const float* b2,
                              float* __restrict__ out, int nNodes) {
  __shared__ float sA[16 * LDSW];
  __shared__ float sB[16 * LDSW];
  int lane = threadIdx.x, tile = blockIdx.x;
  int m    = lane & 15;
  int node = tile * 16 + m;
  bool valid = (node < nNodes);
  if (!valid) node = nNodes - 1;
  const __bf16* hrow = h_bf + (size_t)node * HID;

  v8f zero = {};
  v8f acc[8];
#pragma unroll
  for (int nt = 0; nt < 8; ++nt) acc[nt] = zero;
  for (int kt = 0; kt < 4; ++kt) {
    v16bf a = a_from_bf16row(hrow, kt, lane);
#pragma unroll
    for (int nt = 0; nt < 8; ++nt)
      acc[nt] = wmma_bf16(a, load_b(Wp0, kt, nt, 8, lane), acc[nt]);
  }
  store_tile(sA, acc, b0, lane, true);
  LDS_SYNC();
  dense_from_lds(sA, sB, Wp1, b1, 4, lane, true);
  LDS_SYNC();

  // final 128 -> 2 (N padded to 16 in the packed weights)
  v8f acc2 = zero;
  const float* row = sB + (size_t)m * LDSW;
  for (int kt = 0; kt < 4; ++kt) {
    v16bf a = a_from_f32row(row, kt, lane);
    acc2 = wmma_bf16(a, load_b(Wp2, kt, 0, 1, lane), acc2);
  }
  int col = lane & 15;
  int rb  = (lane & 16) ? 8 : 0;
  if (col < 2) {
#pragma unroll
    for (int r = 0; r < 8; ++r) {
      int n = tile * 16 + rb + r;
      if (n < nNodes) out[(size_t)n * 2 + col] = acc2[r] + b2[col];
    }
  }
}

// ---------------------------------------------------------------------------
// Host-side orchestration
// ---------------------------------------------------------------------------
extern "C" void kernel_launch(void* const* d_in, const int* in_sizes, int n_in,
                              void* d_out, int out_size, void* d_ws, size_t ws_size,
                              hipStream_t stream) {
  // Input order: particle_type, node_attr, edge_index, edge_attr, node_dist,
  // then params flattened in sorted pytree-key order (see analysis).
  const int IDX_PT = 0, IDX_NATTR = 1, IDX_EIDX = 2, IDX_EATTR = 3, IDX_NDIST = 4;
  const int IDX_EDGEIN = 5;    // W0,b0,W1,b1,W2,b2,ln.b,ln.g
  const int IDX_EMBED  = 13;
  const int IDX_LAYERS = 14;   // 16 leaves per MP layer (lin_edge then lin_node)
  const int IDX_NODEIN = 174;
  const int IDX_NODEOUT = 182; // W0,b0,W1,b1,W2,b2
  if (n_in < 188) return;

  const int N = in_sizes[IDX_PT];     // 50000
  const int E = in_sizes[IDX_NDIST];  // 500000

  auto P = [&](int i) { return (const float*)d_in[i]; };

  // Workspace carve-out (256B aligned)
  size_t off = 0;
  auto carve = [&](size_t bytes) -> void* {
    off = (off + 255) & ~(size_t)255;
    void* p = (char*)d_ws + off;
    off += bytes;
    return p;
  };
  float*  h_f32 = (float*) carve((size_t)N * HID * 4);
  __bf16* h_bf  = (__bf16*)carve((size_t)N * HID * 2);
  float*  aggr  = (float*) carve((size_t)N * HID * 4);
  __bf16* e_bf  = (__bf16*)carve((size_t)E * HID * 2);
  __bf16* wbase = (__bf16*)carve((size_t)4 * 1024 * 1024);
  if (off > ws_size) return;  // not enough scratch: deterministic no-op

  // Pack all weight matrices into WMMA-B layout (bf16)
  __bf16* wcur = wbase;
  auto packW = [&](int idx, int K, int Nw) -> const __bf16* {
    int Kt = (K + 31) / 32, Nt = (Nw + 15) / 16, tiles = Kt * Nt;
    __bf16* dstp = wcur;
    int threads = tiles * 32;
    pack_w_kernel<<<(threads + 255) / 256, 256, 0, stream>>>(
        (const float*)d_in[idx], dstp, K, Nw, tiles, Nt);
    wcur += (size_t)tiles * 512;
    return dstp;
  };

  const __bf16 *eiW0 = packW(IDX_EDGEIN + 0, 3, 128),
               *eiW1 = packW(IDX_EDGEIN + 2, 128, 128),
               *eiW2 = packW(IDX_EDGEIN + 4, 128, 128);
  const __bf16 *niW0 = packW(IDX_NODEIN + 0, 30, 128),
               *niW1 = packW(IDX_NODEIN + 2, 128, 128),
               *niW2 = packW(IDX_NODEIN + 4, 128, 128);
  const __bf16 *leW0[10], *leW1[10], *leW2[10], *lnW0[10], *lnW1[10], *lnW2[10];
  for (int i = 0; i < 10; ++i) {
    int b = IDX_LAYERS + 16 * i;
    leW0[i] = packW(b + 0, 384, 128);
    leW1[i] = packW(b + 2, 128, 128);
    leW2[i] = packW(b + 4, 128, 128);
    lnW0[i] = packW(b + 8, 256, 128);
    lnW1[i] = packW(b + 10, 128, 128);
    lnW2[i] = packW(b + 12, 128, 128);
  }
  const __bf16 *noW0 = packW(IDX_NODEOUT + 0, 128, 128),
               *noW1 = packW(IDX_NODEOUT + 2, 128, 128),
               *noW2 = packW(IDX_NODEOUT + 4, 128, 2);

  // Zero aggregation buffer once (node_update re-zeroes it every layer)
  {
    long n = (long)N * HID;
    zero_f32_kernel<<<(unsigned)((n + 255) / 256), 256, 0, stream>>>(aggr, n);
  }

  int nTiles = (N + 15) / 16, eTiles = (E + 15) / 16;

  // Encoders
  encode_edges_kernel<<<eTiles, 32, 0, stream>>>(
      P(IDX_EATTR), eiW0, eiW1, eiW2,
      P(IDX_EDGEIN + 1), P(IDX_EDGEIN + 3), P(IDX_EDGEIN + 5),
      P(IDX_EDGEIN + 7), P(IDX_EDGEIN + 6), e_bf, E);
  encode_nodes_kernel<<<nTiles, 32, 0, stream>>>(
      (const int*)d_in[IDX_PT], P(IDX_NATTR), P(IDX_EMBED),
      niW0, niW1, niW2,
      P(IDX_NODEIN + 1), P(IDX_NODEIN + 3), P(IDX_NODEIN + 5),
      P(IDX_NODEIN + 7), P(IDX_NODEIN + 6), h_f32, h_bf, N);

  // 10 message-passing layers
  for (int i = 0; i < 10; ++i) {
    int b = IDX_LAYERS + 16 * i;
    edge_msg_kernel<<<eTiles, 32, 0, stream>>>(
        h_bf, e_bf, aggr, (const int*)d_in[IDX_EIDX], P(IDX_NDIST),
        leW0[i], leW1[i], leW2[i],
        P(b + 1), P(b + 3), P(b + 5), P(b + 7), P(b + 6), E);
    node_update_kernel<<<nTiles, 32, 0, stream>>>(
        h_f32, h_bf, aggr,
        lnW0[i], lnW1[i], lnW2[i],
        P(b + 9), P(b + 11), P(b + 13), P(b + 15), P(b + 14), N);
  }

  // Decoder
  decode_kernel<<<nTiles, 32, 0, stream>>>(
      h_bf, noW0, noW1, noW2,
      P(IDX_NODEOUT + 1), P(IDX_NODEOUT + 3), P(IDX_NODEOUT + 5),
      (float*)d_out, N);
  (void)out_size;
}
